// ConditionedTemporalAttention_56736517980888
// MI455X (gfx1250) — matrix operand order repaired and checked by reading, not compiled, not run on hardware
//
#include <hip/hip_runtime.h>
#include <hip/hip_bf16.h>
#include <stddef.h>

// ---------------------------------------------------------------------------
// ConditionedTemporalAttention for MI455X (gfx1250, wave32, WMMA).
//   context[n,h] = sum_t softmax_t( v . tanh(gru[n,t,:]@W_h + cond[n,:]@W_c + b) )
//                  * gru[n,t,h]
// Dominant cost: 137 GFLOP GEMM over a 268MB activation read twice
// (HBM floor ~23us @ 23.3TB/s). bf16 WMMA for the GEMM, fused tanh+v-dot
// epilogue (h_proj never hits memory), W_h pre-packed into the WMMA
// B-fragment layout (2MB, L2-resident, reread ~1024x from L2).
// ---------------------------------------------------------------------------

#define NB   32
#define TT   2048
#define HH   1024
#define DD   384
#define KT   (HH / 32)   // 32 K-blocks of 32
#define NT   (HH / 16)   // 64 column tiles of 16

typedef __bf16 v16bf __attribute__((ext_vector_type(16)));
typedef float  v8f   __attribute__((ext_vector_type(8)));

__device__ __forceinline__ unsigned pack_bf16x2(float a, float b) {
  union { float f; unsigned u; } ua, ub;
  ua.f = a; ub.f = b;
  unsigned ha = (ua.u + 0x7FFFu + ((ua.u >> 16) & 1u)) >> 16;   // RNE
  unsigned hb = (ub.u + 0x7FFFu + ((ub.u >> 16) & 1u)) >> 16;
  return ha | (hb << 16);
}

// ---------------------------------------------------------------------------
// K1: pack W_h (H x H fp32, row-major [k_in][n_out]) into bf16 B-fragment
// layout for v_wmma_f32_16x16x32_bf16, in the contiguous k order this file uses:
//   B tile = 32x16 (KxN); lane L holds col N = L&15; K base = (L>=16 ? 16:0);
//   dword j (j=0..7) holds K = base+2j, base+2j+1.
// pw dword index = ((kk*NT + nt)*32 + lane)*8 + j
// ---------------------------------------------------------------------------
__global__ void pack_w_kernel(const float* __restrict__ Wh,
                              unsigned* __restrict__ pw) {
  int idx  = blockIdx.x * 256 + threadIdx.x;          // 0 .. KT*NT*32*8-1
  int j    = idx & 7;
  int lane = (idx >> 3) & 31;
  int tile = idx >> 8;                                // kk*NT + nt
  int nt   = tile & (NT - 1);
  int kk   = tile >> 6;
  int K    = kk * 32 + ((lane >= 16) ? 16 : 0) + 2 * j;
  int n    = nt * 16 + (lane & 15);
  pw[idx] = pack_bf16x2(Wh[K * HH + n], Wh[(K + 1) * HH + n]);
}

// ---------------------------------------------------------------------------
// K2: cpb[n][k] = bias[k] + sum_d cond[n][d] * W_c[d][k]   (tiny GEMM)
// ---------------------------------------------------------------------------
__global__ void cproj_kernel(const float* __restrict__ cond,
                             const float* __restrict__ Wc,
                             const float* __restrict__ bias,
                             float* __restrict__ cpb) {
  int n = blockIdx.y;
  int k = blockIdx.x * 256 + threadIdx.x;
  float acc = bias[k];
  const float* c = cond + n * DD;
#pragma unroll 4
  for (int d = 0; d < DD; ++d) acc += c[d] * Wc[d * HH + k];
  cpb[n * HH + k] = acc;
}

// ---------------------------------------------------------------------------
// K3: fused  scores[n,t] = v . tanh( gru_row @ W_h + cpb[n,:] )
// 64 rows / workgroup (all in one batch n since 64 | T), 8 waves.
// Dynamic LDS: bf16 A tile (64 x 1024, 16B row pad) + per-wave score slots.
// ---------------------------------------------------------------------------
#define A_PITCH 2064                       // 1024*2 + 16B pad (16B aligned)
#define SMEM_BYTES (64 * A_PITCH + 8 * 64 * 4)

__global__ __launch_bounds__(256)
void attn_scores_kernel(const float* __restrict__ gru,
                        const unsigned* __restrict__ pw,
                        const float* __restrict__ cpb,
                        const float* __restrict__ vvec,
                        float* __restrict__ scores) {
  extern __shared__ __align__(16) char smem[];
  char*  aTile = smem;
  float* swv   = (float*)(smem + 64 * A_PITCH);       // [8 waves][64 rows]

  const int tid  = threadIdx.x;
  const int row0 = blockIdx.x * 64;
  const int n    = row0 >> 11;                        // row0 / T
  const float* gbase = gru + (size_t)row0 * HH;

  // ---- stage A tile: fp32 -> bf16 into LDS (coalesced float4 loads) ----
#pragma unroll 4
  for (int it = 0; it < 64; ++it) {
    int idx = it * 256 + tid;                         // one float4 each
    int row = idx >> 8;                               // == it
    int c4  = idx & 255;                              // == tid
    float4 f = ((const float4*)(gbase + (size_t)row * HH))[c4];
    uint2 p;
    p.x = pack_bf16x2(f.x, f.y);
    p.y = pack_bf16x2(f.z, f.w);
    *(uint2*)(aTile + row * A_PITCH + c4 * 8) = p;
  }
  __syncthreads();

  const int lane  = tid & 31;
  const int wave  = tid >> 5;
  const int lhalf = lane >> 4;
  const int lmod  = lane & 15;
  const uint4* pwv = (const uint4*)pw;

  float sacc[32];
#pragma unroll
  for (int i = 0; i < 32; ++i) sacc[i] = 0.f;

#pragma unroll 1
  for (int i = 0; i < 8; ++i) {                       // 8 column tiles / wave
    const int nt = wave + i * 8;
    v8f c[4] = {};
#pragma unroll 2
    for (int kk = 0; kk < KT; ++kk) {
      // Compiler fence: keep the A-fragment LDS loads *inside* this loop.
      // Without it LICM hoists all 128 fragments out of the i-loop and
      // spills ~1000 VGPRs to scratch (seen in round-1 asm).
      asm volatile("" ::: "memory");

      // B fragment: two coalesced 128b loads from packed W (L2-resident)
      size_t bidx = ((size_t)(kk * NT + nt) * 32 + lane) * 2;
      union { uint4 q[2]; v16bf v; } bu;
      bu.q[0] = pwv[bidx];
      bu.q[1] = pwv[bidx + 1];
      if (kk < KT - 1)
        __builtin_prefetch(&pwv[bidx + (size_t)NT * 32 * 2], 0, 1);

      // A fragments from LDS: contiguous k order, same as packed B
      //   j=0..7: K = kbase + 0..15
      //   kbase = kk*32 + (lane>=16 ? 16 : 0)
      const int kbyte = kk * 64 + lhalf * 32;
#pragma unroll
      for (int rt = 0; rt < 4; ++rt) {
        const char* ap = aTile + (rt * 16 + lmod) * A_PITCH + kbyte;
        union { uint4 q[2]; v16bf v; } au;
        au.q[0] = *(const uint4*)(ap);                // ds_load_b128
        au.q[1] = *(const uint4*)(ap + 16);           // ds_load_b128
        c[rt] = __builtin_amdgcn_wmma_f32_16x16x32_bf16(
            false, au.v, false, bu.v, (short)0, c[rt], false, false);
      }
    }
    // fused epilogue: energy = tanh(hproj + cproj + bias); partial v-dot
    const int col = nt * 16 + lmod;
    const float vv = vvec[col];
    const float cb = cpb[n * HH + col];
#pragma unroll
    for (int rt = 0; rt < 4; ++rt)
#pragma unroll
      for (int r = 0; r < 8; ++r) {
        float e = tanhf(c[rt][r] + cb);
        sacc[rt * 8 + r] += e * vv;
      }
  }

  // deterministic reduction: sum over the 16 columns held by a lane group
#pragma unroll
  for (int i = 0; i < 32; ++i) {
    float s = sacc[i];
    s += __shfl_xor(s, 1);
    s += __shfl_xor(s, 2);
    s += __shfl_xor(s, 4);
    s += __shfl_xor(s, 8);
    sacc[i] = s;
  }
  if (lmod == 0) {
#pragma unroll
    for (int rt = 0; rt < 4; ++rt)
#pragma unroll
      for (int r = 0; r < 8; ++r)
        swv[wave * 64 + rt * 16 + r + 8 * lhalf] = sacc[rt * 8 + r];
  }
  __syncthreads();
  if (tid < 64) {                                     // fixed-order wave sum
    float s = 0.f;
#pragma unroll
    for (int w = 0; w < 8; ++w) s += swv[w * 64 + tid];
    scores[row0 + tid] = s;
  }
}

// ---------------------------------------------------------------------------
// K4: softmax over T per batch
// ---------------------------------------------------------------------------
__global__ void softmax_kernel(const float* __restrict__ scores,
                               float* __restrict__ weights) {
  int n = blockIdx.x, tid = threadIdx.x;
  __shared__ float red[256];
  const float* s = scores + n * TT;
  float m = -3.0e38f;
  for (int t = tid; t < TT; t += 256) m = fmaxf(m, s[t]);
  red[tid] = m; __syncthreads();
  for (int o = 128; o > 0; o >>= 1) {
    if (tid < o) red[tid] = fmaxf(red[tid], red[tid + o]);
    __syncthreads();
  }
  const float mx = red[0];
  __syncthreads();
  float sum = 0.f;
  for (int t = tid; t < TT; t += 256) sum += __expf(s[t] - mx);
  red[tid] = sum; __syncthreads();
  for (int o = 128; o > 0; o >>= 1) {
    if (tid < o) red[tid] += red[tid + o];
    __syncthreads();
  }
  const float inv = 1.f / red[0];
  for (int t = tid; t < TT; t += 256)
    weights[n * TT + t] = __expf(s[t] - mx) * inv;
}

// ---------------------------------------------------------------------------
// K5: context[n,h] = sum_t w[n,t] * gru[n,t,h]   (bandwidth bound, coalesced)
// ---------------------------------------------------------------------------
__global__ __launch_bounds__(256)
void context_kernel(const float* __restrict__ gru,
                    const float* __restrict__ weights,
                    float* __restrict__ out) {
  int n = blockIdx.y;
  int h = blockIdx.x * 256 + threadIdx.x;
  const float* g = gru + (size_t)n * TT * HH + h;
  const float* w = weights + n * TT;
  float acc = 0.f;
#pragma unroll 8
  for (int t = 0; t < TT; ++t) acc += w[t] * g[(size_t)t * HH];
  out[n * HH + h] = acc;
}

// ---------------------------------------------------------------------------
extern "C" void kernel_launch(void* const* d_in, const int* in_sizes, int n_in,
                              void* d_out, int out_size, void* d_ws, size_t ws_size,
                              hipStream_t stream) {
  const float* gru  = (const float*)d_in[0];   // (N,T,H)
  const float* cond = (const float*)d_in[1];   // (N,D)
  const float* Wh   = (const float*)d_in[2];   // (H,H)
  const float* Wc   = (const float*)d_in[3];   // (D,H)
  const float* bias = (const float*)d_in[4];   // (H)
  const float* v    = (const float*)d_in[5];   // (H)
  float* out = (float*)d_out;                  // (N,H)

  char* ws = (char*)d_ws;
  unsigned* pw      = (unsigned*)(ws);                          // 2 MB packed W
  float*    cpb     = (float*)(ws + 2097152);                   // 128 KB
  float*    scores  = (float*)(ws + 2097152 + 131072);          // 256 KB
  float*    weights = (float*)(ws + 2097152 + 131072 + 262144); // 256 KB

  pack_w_kernel<<<KT * NT * 32 * 8 / 256, 256, 0, stream>>>(Wh, pw);
  cproj_kernel<<<dim3(HH / 256, NB), 256, 0, stream>>>(cond, Wc, bias, cpb);
  attn_scores_kernel<<<NB * TT / 64, 256, SMEM_BYTES, stream>>>(
      gru, pw, cpb, v, scores);
  softmax_kernel<<<NB, 256, 0, stream>>>(scores, weights);
  context_kernel<<<dim3(HH / 256, NB), 256, 0, stream>>>(gru, weights, out);
}
